// Encoder_11184094838958
// MI455X (gfx1250) — matrix operand order repaired and checked by reading, not compiled, not run on hardware
//
#include <hip/hip_runtime.h>
#include <hip/hip_bf16.h>
#include <math.h>

// Problem constants (match reference)
#define TT 2048
#define BB 64
#define FIN 88
#define HH 128      // hidden
#define GG 512      // 4*H
#define KPH 136     // padded row stride (halves) for h in LDS: 272B = 17*16B -> aligned, bank-staggered

typedef __attribute__((ext_vector_type(16))) __bf16 v16bf;
typedef __attribute__((ext_vector_type(8)))  float  v8f;

// ---------------- WMMA fragment loaders (wave32, 16x16x32 bf16) ----------------
// A (16x32, 16-bit): lane L holds row M=L&15; halves 0..15 = K kb..kb+15 contiguous,
// kb = (L>>4)*16.  Requires 16B alignment of per-lane pointers.
__device__ __forceinline__ v16bf frag_a(const __bf16* base, int ld, int lane) {
  const __bf16* p = base + (lane & 15) * ld + ((lane >> 4) << 4);
  union { v16bf v; uint4 q[2]; } u;
  u.q[0] = *(const uint4*)(p);
  u.q[1] = *(const uint4*)(p + 8);
  return u.v;
}
// B (32x16, 16-bit) from row-major transposed weights Wt[N][K]: lane L holds column N=n0+(L&15),
// halves = K kst..kst+15 contiguous, kst = (L>>4)*16.
__device__ __forceinline__ v16bf frag_b_lin(const __bf16* baseT, int ld, int lane) {
  const __bf16* p = baseT + (lane & 15) * ld + ((lane >> 4) << 4);
  union { v16bf v; uint4 q[2]; } u;
  u.q[0] = *(const uint4*)(p);
  u.q[1] = *(const uint4*)(p + 8);
  return u.v;
}
// B from pre-swizzled fragment storage: 512 halves per fragment, lane-major 16-half chunks.
__device__ __forceinline__ v16bf frag_b_sw(const __bf16* fragbase, int lane) {
  const __bf16* p = fragbase + (lane << 4);
  union { v16bf v; uint4 q[2]; } u;
  u.q[0] = *(const uint4*)(p);
  u.q[1] = *(const uint4*)(p + 8);
  return u.v;
}

__device__ __forceinline__ float sigf(float x) { return 1.f / (1.f + __expf(-x)); }

// ---------------- Kernel 1: lift GEMM + transpose to time-major bf16 ----------------
// xt[t][b][h] = inputs[b][t][:88] @ lift_w[:,h] + lift_b[h]
__global__ __launch_bounds__(256) void lift_kernel(const float* __restrict__ inp,
                                                   const float* __restrict__ lw,
                                                   const float* __restrict__ lb,
                                                   __bf16* __restrict__ xt) {
  __shared__ __bf16 Aa[64 * 96];     // [b][k], K padded 88->96 (zero fill)
  __shared__ __bf16 Bt[128 * 96];    // lift_w transposed [n][k]
  __shared__ float  lbias[128];
  const int tid = threadIdx.x;
  const int t = blockIdx.x;

  for (int idx = tid; idx < 64 * 96; idx += 256) {
    int b = idx / 96, f = idx % 96;
    Aa[idx] = (f < FIN) ? (__bf16)inp[((size_t)b * TT + t) * FIN + f] : (__bf16)0.f;
  }
  for (int idx = tid; idx < 128 * 96; idx += 256) {
    int n = idx / 96, k = idx % 96;
    Bt[idx] = (k < FIN) ? (__bf16)lw[k * HH + n] : (__bf16)0.f;
  }
  if (tid < 128) lbias[tid] = lb[tid];
  __syncthreads();

  const int wid = tid >> 5, lane = tid & 31;
  #pragma unroll
  for (int tt = 0; tt < 4; ++tt) {
    const int tau = wid + (tt << 3);      // 0..31
    const int mt = tau >> 3, nt = tau & 7;
    v8f acc = {};
    #pragma unroll
    for (int kc = 0; kc < 3; ++kc) {
      v16bf a = frag_a(Aa + (mt << 4) * 96 + (kc << 5), 96, lane);
      v16bf b = frag_b_lin(Bt + (nt << 4) * 96 + (kc << 5), 96, lane);
      acc = __builtin_amdgcn_wmma_f32_16x16x32_bf16(false, a, false, b, (short)0, acc, false, false);
    }
    const int n = (nt << 4) + (lane & 15);
    const int rowb = (mt << 4) + ((lane >> 4) << 3);
    #pragma unroll
    for (int r = 0; r < 8; ++r)
      xt[((size_t)t * BB + rowb + r) * HH + n] = (__bf16)(acc[r] + lbias[n]);
  }
}

// ---------------- Scan kernel helpers ----------------
// One z half-pass: compute z[:, half*256 .. half*256+255] = [x_t ; h] @ [W ; U] for 64 rows.
// 4 M-tiles x 16 N-tiles = 64 tiles, 32 waves -> 2 tiles/wave, 8 K-chunks (4 from W/x, 4 from U/h).
__device__ __forceinline__ void zpass(const __bf16* __restrict__ xs, const __bf16* hb,
                                      const __bf16* WB, const __bf16* UB, __bf16* zh,
                                      int half, int wid, int lane) {
  #pragma unroll
  for (int tt = 0; tt < 2; ++tt) {
    const int tau = wid + (tt << 5);
    const int mt = tau >> 4, nt = tau & 15;
    const int ntg = (half << 4) + nt;       // global N-tile 0..31
    v8f acc = {};
    const __bf16* xrow = xs + (mt << 4) * HH;
    const __bf16* hrow = hb + (mt << 4) * KPH;
    #pragma unroll
    for (int kc = 0; kc < 4; ++kc) {        // x @ W part (K 0..127)
      v16bf a = frag_a(xrow + (kc << 5), HH, lane);
      v16bf b = frag_b_sw(WB + (size_t)(((ntg << 2) + kc) << 9), lane);
      acc = __builtin_amdgcn_wmma_f32_16x16x32_bf16(false, a, false, b, (short)0, acc, false, false);
    }
    #pragma unroll
    for (int kc = 0; kc < 4; ++kc) {        // h @ U part (K 128..255)
      v16bf a = frag_a(hrow + (kc << 5), KPH, lane);
      v16bf b = frag_b_sw(UB + (size_t)(((ntg << 2) + kc) << 9), lane);
      acc = __builtin_amdgcn_wmma_f32_16x16x32_bf16(false, a, false, b, (short)0, acc, false, false);
    }
    const int colb = (nt << 4) + (lane & 15);
    const int rowb = (mt << 4) + ((lane >> 4) << 3);
    #pragma unroll
    for (int r = 0; r < 8; ++r) zh[(rowb + r) * 256 + colb] = (__bf16)acc[r];
  }
}

// ---------------- Kernel 2: persistent LSTM scan (one block per direction) ----------------
// LDS: WB 128K | UB 128K | h 17K | z-half 32K | bias 2K  = 314,368 B  (<= 320KB/WGP)
#define SMEM_BYTES 314368

__global__ __launch_bounds__(1024) void scan_kernel(
    const __bf16* __restrict__ xin_fw, const __bf16* __restrict__ xin_bw,
    const float* __restrict__ Wf, const float* __restrict__ Uf, const float* __restrict__ Bf,
    const float* __restrict__ Wb, const float* __restrict__ Ub, const float* __restrict__ Bb,
    __bf16* __restrict__ hseq_fw, __bf16* __restrict__ hseq_bw,
    float* __restrict__ out, float* __restrict__ cstate, int layer) {
  extern __shared__ char smem[];
  __bf16* WB   = (__bf16*)(smem);
  __bf16* UB   = (__bf16*)(smem + 131072);
  __bf16* hb   = (__bf16*)(smem + 262144);
  __bf16* zh   = (__bf16*)(smem + 279552);
  float*  bias = (float*)(smem + 312320);

  const int tid = threadIdx.x;
  const int dir = blockIdx.x;                 // 0 = fw, 1 = bw
  const __bf16* xin = dir ? xin_bw : xin_fw;
  const float* Wg = dir ? Wb : Wf;
  const float* Ug = dir ? Ub : Uf;
  const float* Bg = dir ? Bb : Bf;
  __bf16* hseq = dir ? hseq_bw : hseq_fw;

  // Load weights into LDS, pre-swizzled into WMMA-B fragment order:
  // fragment f = nt*4+kc holds 32 lanes x 16 contiguous K halves.
  for (int idx = tid; idx < 65536; idx += 1024) {
    const int f = idx >> 9, l9 = (idx >> 4) & 31, i = idx & 15;
    const int nt = f >> 2, kc = f & 3;
    const int n = (nt << 4) + (l9 & 15);
    const int k = (kc << 5) + ((l9 >> 4) << 4) + i;
    WB[idx] = (__bf16)Wg[k * GG + n];
    UB[idx] = (__bf16)Ug[k * GG + n];
  }
  if (tid < GG) bias[tid] = Bg[tid];
  for (int idx = tid; idx < BB * KPH; idx += 1024) hb[idx] = (__bf16)0.f;

  // Per-thread gate ownership: 8 consecutive h-elements of one batch row.
  const int eb = tid << 3;
  const int be = eb >> 7;        // batch index 0..63
  const int he = eb & 127;       // h offset, multiple of 8
  float c[8];
  #pragma unroll
  for (int j = 0; j < 8; ++j) c[j] = 0.f;
  float si[8], sf[8];

  __syncthreads();
  const int wid = tid >> 5, lane = tid & 31;

  for (int s = 0; s < TT; ++s) {
    const int sidx = (layer == 0 && dir == 1) ? (TT - 1 - s) : s;
    const __bf16* xs = xin + (size_t)sidx * (BB * HH);

    // pass 1: gates i (cols 0..127), f (128..255)
    zpass(xs, hb, WB, UB, zh, 0, wid, lane);
    __syncthreads();
    {
      const __bf16* zi = zh + be * 256 + he;
      #pragma unroll
      for (int j = 0; j < 8; ++j) {
        si[j] = sigf((float)zi[j]       + bias[he + j]);
        sf[j] = sigf((float)zi[128 + j] + bias[128 + he + j]);
      }
      if (s + 1 < TT) {  // prefetch next step's x tile (global_prefetch_b8)
        const int nidx = (layer == 0 && dir == 1) ? (TT - 2 - s) : (s + 1);
        __builtin_prefetch(xin + (size_t)nidx * (BB * HH) + (tid << 3), 0, 0);
      }
    }
    __syncthreads();
    // pass 2: gates g (256..383), o (384..511), reuse zh
    zpass(xs, hb, WB, UB, zh, 1, wid, lane);
    __syncthreads();
    {
      const __bf16* zg = zh + be * 256 + he;
      #pragma unroll
      for (int j = 0; j < 8; ++j) {
        const float g = (float)zg[j]       + bias[256 + he + j];
        const float o = (float)zg[128 + j] + bias[384 + he + j];
        c[j] = sf[j] * c[j] + si[j] * tanhf(g);
        const float hv = sigf(o) * tanhf(c[j]);
        hb[be * KPH + he + j] = (__bf16)hv;
        if (layer == 0) {
          hseq[((size_t)s * BB + be) * HH + he + j] = (__bf16)hv;
        } else {
          const int t = dir ? (TT - 1 - s) : s;
          out[(size_t)be * (TT * 256) + (size_t)t * 256 + (dir << 7) + he + j] = hv;
        }
      }
    }
    __syncthreads();
  }

  // final_state = concat([c0_fw, c1_fw], axis=1) -> [64, 256]
  if (dir == 0) {
    #pragma unroll
    for (int j = 0; j < 8; ++j)
      cstate[be * 256 + (layer << 7) + he + j] = c[j];
  }
}

// ---------------- host launcher ----------------
extern "C" void kernel_launch(void* const* d_in, const int* in_sizes, int n_in,
                              void* d_out, int out_size, void* d_ws, size_t ws_size,
                              hipStream_t stream) {
  (void)in_sizes; (void)n_in; (void)out_size; (void)ws_size;
  const float* inputs = (const float*)d_in[0];
  const float* lift_w = (const float*)d_in[1];
  const float* lift_b = (const float*)d_in[2];
  const float* w_fw0 = (const float*)d_in[3];
  const float* u_fw0 = (const float*)d_in[4];
  const float* b_fw0 = (const float*)d_in[5];
  const float* w_fw1 = (const float*)d_in[6];
  const float* u_fw1 = (const float*)d_in[7];
  const float* b_fw1 = (const float*)d_in[8];
  const float* w_bw0 = (const float*)d_in[9];
  const float* u_bw0 = (const float*)d_in[10];
  const float* b_bw0 = (const float*)d_in[11];
  const float* w_bw1 = (const float*)d_in[12];
  const float* u_bw1 = (const float*)d_in[13];
  const float* b_bw1 = (const float*)d_in[14];

  char* ws = (char*)d_ws;
  const size_t SEQ_BYTES = (size_t)TT * BB * HH * sizeof(__bf16);  // 32 MB
  __bf16* xt  = (__bf16*)(ws);                  // lifted input, time-major bf16
  __bf16* hfw = (__bf16*)(ws + SEQ_BYTES);      // fw layer0 h sequence
  __bf16* hbw = (__bf16*)(ws + 2 * SEQ_BYTES);  // bw layer0 h sequence (reversed order)

  float* out = (float*)d_out;
  float* cstate = out + (size_t)BB * TT * 256;  // after outputs [64,2048,256]

  hipFuncSetAttribute(reinterpret_cast<const void*>(&scan_kernel),
                      hipFuncAttributeMaxDynamicSharedMemorySize, SMEM_BYTES);

  lift_kernel<<<TT, 256, 0, stream>>>(inputs, lift_w, lift_b, xt);

  // layer 0: fw + bw in parallel (block 0 / block 1)
  scan_kernel<<<2, 1024, SMEM_BYTES, stream>>>(xt, xt,
      w_fw0, u_fw0, b_fw0, w_bw0, u_bw0, b_bw0,
      hfw, hbw, out, cstate, 0);

  // layer 1: consumes layer-0 h sequences, writes outputs + c1_fw
  scan_kernel<<<2, 1024, SMEM_BYTES, stream>>>(hfw, hbw,
      w_fw1, u_fw1, b_fw1, w_bw1, u_bw1, b_bw1,
      nullptr, nullptr, out, cstate, 1);
}
